// MultiHeadAttention_88931592831377
// MI455X (gfx1250) — matrix operand order repaired and checked by reading, not compiled, not run on hardware
//
#include <hip/hip_runtime.h>
#include <hip/hip_bf16.h>

// ---------------------------------------------------------------------------
// CDNA5 (gfx1250) wave32 WMMA multi-head attention
//   B=2, S=2048, D=2048, H=16, hd=128
// Pipeline: [gemm_proj x3] -> [rope] -> [flash attention] -> [gemm_out]
// Matrix math via v_wmma_f32_16x16x32_bf16; bf16 operand staging via
// GLOBAL_LOAD_ASYNC_TO_LDS_B128 (ASYNCcnt) where no conversion is needed.
// ---------------------------------------------------------------------------

typedef __bf16 bf16_t;
typedef __attribute__((ext_vector_type(16))) __bf16 v16bf;
typedef __attribute__((ext_vector_type(8)))  __bf16 v8bf;
typedef __attribute__((ext_vector_type(8)))  float  v8f;

#define WMMA_BF16(A, Bm, C) \
  __builtin_amdgcn_wmma_f32_16x16x32_bf16(false, (A), false, (Bm), (short)0, (C), false, false)

// Async global->LDS copy, 16 bytes per lane (ISA 10. / 15.18.3 opcode 98).
// LDS destination offset comes from the low 32 bits of the flat address
// (via flat-to-LDS aperture translation).
#define ASYNC_G2L_B128(ldsOff, gaddr)                                    \
  asm volatile("global_load_async_to_lds_b128 %0, %1, off"               \
               :: "v"(ldsOff), "v"(gaddr) : "memory")
#define WAIT_ASYNC0() asm volatile("s_wait_asynccnt 0" ::: "memory")

static constexpr int kB   = 2;
static constexpr int kS   = 2048;
static constexpr int kD   = 2048;
static constexpr int kH   = 16;
static constexpr int kHd  = 128;
static constexpr int kM   = kB * kS;      // 4096 rows of x

__device__ __forceinline__ unsigned lds_off32(const void* p) {
  return (unsigned)(uintptr_t)p;          // low 32 bits of flat LDS address
}

// ---------------------------------------------------------------------------
// Fragment loaders (ISA 7.12.2 wave32 layouts).
// A (16x32 MxK, 16-bit): lanes 0-15 hold M=0..15 with K chunks {0..7,16..23};
//                        lanes 16-31 hold M=0..15 with K chunks {8..15,24..31}.
// B (32x16 KxN, 16-bit): lanes 0-15 hold N=0..15 with K chunks {0..7,16..23};
//                        lanes 16-31 hold N=0..15 with K chunks {8..15,24..31}.
// LDS strides are multiples of 8 elements so every v8bf access is 16B aligned.
// ---------------------------------------------------------------------------
__device__ __forceinline__ v16bf load_frag_a(const bf16_t* base, int ld,
                                             int row, int k0, int lane) {
  const int r  = row + (lane & 15);
  const int kb = k0 + ((lane >> 4) << 3);     // +8 for upper half-wave
  const bf16_t* p = base + r * ld + kb;
  v8bf lo = *(const v8bf*)p;                  // K = kb .. kb+7
  v8bf hi = *(const v8bf*)(p + 16);           // K = kb+16 .. kb+23
  v16bf out;
#pragma unroll
  for (int i = 0; i < 8; ++i) { out[i] = lo[i]; out[8 + i] = hi[i]; }
  return out;
}

__device__ __forceinline__ v16bf load_frag_b(const bf16_t* base, int ld,
                                             int col, int k0, int lane) {
  const int c  = col + (lane & 15);
  const int kb = k0 + ((lane >> 4) << 3);     // +8 for upper half-wave
  const bf16_t* p = base + c * ld + kb;       // operand stored N-major, K contiguous
  v8bf lo = *(const v8bf*)p;
  v8bf hi = *(const v8bf*)(p + 16);
  v16bf out;
#pragma unroll
  for (int i = 0; i < 8; ++i) { out[i] = lo[i]; out[8 + i] = hi[i]; }
  return out;
}

// ---------------------------------------------------------------------------
// Kernel 1: Y = X @ W^T, X f32 [M,K], W f32 [N,K]; output bf16 remapped to
// [B,H,S,hd]. Block = 128 threads (4 waves), 64x64 tile, Kt=64.
// f32 inputs need VALU bf16 conversion, so staging stays on the VGPR path.
// ---------------------------------------------------------------------------
__global__ __launch_bounds__(128)
void gemm_proj_kernel(const float* __restrict__ X, const float* __restrict__ W,
                      bf16_t* __restrict__ Y) {
  __shared__ alignas(16) bf16_t Xs[64][72];
  __shared__ alignas(16) bf16_t Ws[64][72];

  const int tid  = threadIdx.x;
  const int lane = tid & 31;
  const int wave = tid >> 5;
  const int m0 = blockIdx.y * 64;
  const int n0 = blockIdx.x * 64;
  const int wm = (wave >> 1) * 32;
  const int wn = (wave & 1) * 32;

  v8f acc[2][2] = {};

  const int lr = tid >> 1;            // staging row 0..63
  const int lc = (tid & 1) * 32;      // staging col base

  for (int k0 = 0; k0 < kD; k0 += 64) {
    const float* xp = X + (size_t)(m0 + lr) * kD + k0 + lc;
    const float* wp = W + (size_t)(n0 + lr) * kD + k0 + lc;
    if (k0 + 64 < kD) {               // hint L2 for the next K tile
      __builtin_prefetch(xp + 64, 0, 1);
      __builtin_prefetch(wp + 64, 0, 1);
    }
#pragma unroll
    for (int c = 0; c < 32; c += 8) {
      v8bf xv, wv;
#pragma unroll
      for (int j = 0; j < 8; ++j) {
        xv[j] = (bf16_t)xp[c + j];
        wv[j] = (bf16_t)wp[c + j];
      }
      *(v8bf*)&Xs[lr][lc + c] = xv;
      *(v8bf*)&Ws[lr][lc + c] = wv;
    }
    __syncthreads();
#pragma unroll
    for (int kk = 0; kk < 64; kk += 32) {
      v16bf a0 = load_frag_a(&Xs[0][0], 72, wm,      kk, lane);
      v16bf a1 = load_frag_a(&Xs[0][0], 72, wm + 16, kk, lane);
      v16bf b0 = load_frag_b(&Ws[0][0], 72, wn,      kk, lane);
      v16bf b1 = load_frag_b(&Ws[0][0], 72, wn + 16, kk, lane);
      acc[0][0] = WMMA_BF16(a0, b0, acc[0][0]);
      acc[0][1] = WMMA_BF16(a0, b1, acc[0][1]);
      acc[1][0] = WMMA_BF16(a1, b0, acc[1][0]);
      acc[1][1] = WMMA_BF16(a1, b1, acc[1][1]);
    }
    __syncthreads();
  }

  // Epilogue: C/D tile layout — lanes 0-15: M=vgpr, N=lane; lanes 16-31: M=vgpr+8.
  const int g   = lane >> 4;
  const int col = lane & 15;
#pragma unroll
  for (int i = 0; i < 2; ++i)
#pragma unroll
    for (int j = 0; j < 2; ++j)
#pragma unroll
      for (int v = 0; v < 8; ++v) {
        const int mrow = m0 + wm + i * 16 + g * 8 + v;
        const int ncol = n0 + wn + j * 16 + col;
        const int b = mrow >> 11, s = mrow & (kS - 1);
        const int h = ncol >> 7,  d = ncol & (kHd - 1);
        Y[(((size_t)b * kH + h) * kS + s) * kHd + d] = (bf16_t)acc[i][j][v];
      }
}

// ---------------------------------------------------------------------------
// Kernel 2: in-place RoPE on Q and K, layout [B,H,S,hd].
// cos/sin tables computed on the fly: inv_freq[i] = 10000^(-2i/hd).
// ---------------------------------------------------------------------------
__global__ __launch_bounds__(256)
void rope_kernel(bf16_t* __restrict__ Q, bf16_t* __restrict__ K) {
  const size_t idx = (size_t)blockIdx.x * 256 + threadIdx.x;  // B*H*S*64
  const int    i   = (int)(idx & 63);
  const size_t row = idx >> 6;                 // b*H*S + h*S + s
  const int    s   = (int)(row & (kS - 1));

  const float freq = __expf(-(float)(2 * i) * (9.210340371976184f / (float)kHd));
  float sn, cs;
  __sincosf((float)s * freq, &sn, &cs);

  const size_t base = row * kHd;
  const float q1 = (float)Q[base + i], q2 = (float)Q[base + i + 64];
  Q[base + i]      = (bf16_t)(q1 * cs - q2 * sn);
  Q[base + i + 64] = (bf16_t)(q2 * cs + q1 * sn);
  const float k1 = (float)K[base + i], k2 = (float)K[base + i + 64];
  K[base + i]      = (bf16_t)(k1 * cs - k2 * sn);
  K[base + i + 64] = (bf16_t)(k2 * cs + k1 * sn);
}

// ---------------------------------------------------------------------------
// Kernel 3: causal flash attention for one (b,h) and a 64-row query tile.
// Block = 128 threads (4 waves); wave w owns query rows [w*16, w*16+16).
// Q and K tiles stream global->LDS via async copies (ASYNCcnt); V is
// transposed through VGPRs (async DMA cannot transpose).
// ---------------------------------------------------------------------------
__global__ __launch_bounds__(128)
void attn_kernel(const bf16_t* __restrict__ Q, const bf16_t* __restrict__ K,
                 const bf16_t* __restrict__ V, bf16_t* __restrict__ Ctx) {
  __shared__ alignas(16) bf16_t Qs[64][136];
  __shared__ alignas(16) bf16_t Ks[64][136];
  __shared__ alignas(16) bf16_t Vt[128][72];   // V transposed: [d][t]
  __shared__ alignas(16) bf16_t Ps[64][72];    // softmax probabilities

  const int tid  = threadIdx.x;
  const int lane = tid & 31;
  const int wave = tid >> 5;
  const int qt   = blockIdx.x;                 // query tile
  const int bh   = blockIdx.y;                 // b*H + h
  const int qbase = qt * 64;

  const bf16_t* Qb = Q + (size_t)bh * kS * kHd;
  const bf16_t* Kb = K + (size_t)bh * kS * kHd;
  const bf16_t* Vb = V + (size_t)bh * kS * kHd;

  const int sr = tid >> 1;            // staging row 0..63
  const int sc = (tid & 1) * 64;      // staging col base

  // Stage the query tile (bf16, 64 x 128) via async DMA.
  {
    const unsigned long long qp =
        (unsigned long long)(uintptr_t)(Qb + (size_t)(qbase + sr) * kHd + sc);
    const unsigned qo = lds_off32(&Qs[sr][sc]);
#pragma unroll
    for (int c = 0; c < 64; c += 8)
      ASYNC_G2L_B128(qo + (unsigned)(c * 2), qp + (unsigned long long)(c * 2));
  }

  const int qr  = wave * 16;
  const int g   = lane >> 4;
  const int col = lane & 15;
  const float scale = 0.08838834764831845f;    // 1/sqrt(128)

  v8f o[8] = {};
  float mi[8], li[8];
#pragma unroll
  for (int v = 0; v < 8; ++v) { mi[v] = -__builtin_inff(); li[v] = 0.f; }

  for (int kt = 0; kt <= qt; ++kt) {
    __syncthreads();                           // prior compute done before restage
    {
      // K tile: async global->LDS.
      const unsigned long long kp =
          (unsigned long long)(uintptr_t)(Kb + (size_t)(kt * 64 + sr) * kHd + sc);
      const unsigned ko = lds_off32(&Ks[sr][sc]);
#pragma unroll
      for (int c = 0; c < 64; c += 8)
        ASYNC_G2L_B128(ko + (unsigned)(c * 2), kp + (unsigned long long)(c * 2));

      // V tile: load + transpose through VGPRs.
      const bf16_t* vp = Vb + (size_t)(kt * 64 + sr) * kHd + sc;
#pragma unroll
      for (int c = 0; c < 64; c += 8) {
        v8bf vv = *(const v8bf*)(vp + c);
#pragma unroll
        for (int j = 0; j < 8; ++j) Vt[sc + c + j][sr] = vv[j];
      }
    }
    WAIT_ASYNC0();                             // Q (first iter) + K in LDS
    __syncthreads();

    // S-tile: Q(16x128) @ K^T(128x64) per wave, K dim in 4 steps of 32.
    v8f sa[4] = {};
#pragma unroll
    for (int ks = 0; ks < 4; ++ks) {
      v16bf a = load_frag_a(&Qs[0][0], 136, qr, ks * 32, lane);
#pragma unroll
      for (int ct = 0; ct < 4; ++ct) {
        v16bf bfrag = load_frag_b(&Ks[0][0], 136, ct * 16, ks * 32, lane);
        sa[ct] = WMMA_BF16(a, bfrag, sa[ct]);
      }
    }

    // Online softmax; row stats live per (vgpr v, half-wave g).
    const int t0g  = kt * 64;
    const bool diag = (kt == qt);
#pragma unroll
    for (int v = 0; v < 8; ++v) {
      const int rowg = qbase + qr + g * 8 + v;
      float mx = -__builtin_inff();
#pragma unroll
      for (int ct = 0; ct < 4; ++ct) {
        float sc2 = sa[ct][v] * scale;
        if (diag && (t0g + ct * 16 + col > rowg)) sc2 = -__builtin_inff();
        sa[ct][v] = sc2;
        mx = fmaxf(mx, sc2);
      }
#pragma unroll
      for (int off = 1; off < 16; off <<= 1) mx = fmaxf(mx, __shfl_xor(mx, off, 32));
      const float mn   = fmaxf(mi[v], mx);
      const float corr = __expf(mi[v] - mn);
      li[v] *= corr;
#pragma unroll
      for (int n = 0; n < 8; ++n) o[n][v] *= corr;
      float ls = 0.f;
#pragma unroll
      for (int ct = 0; ct < 4; ++ct) {
        const float p = __expf(sa[ct][v] - mn);
        Ps[qr + g * 8 + v][ct * 16 + col] = (bf16_t)p;
        ls += p;
      }
#pragma unroll
      for (int off = 1; off < 16; off <<= 1) ls += __shfl_xor(ls, off, 32);
      li[v] += ls;
      mi[v] = mn;
    }

    // Same-wave LDS write -> fragment read hazard on Ps (DScnt, ISA 7.3).
    asm volatile("s_wait_dscnt 0" ::: "memory");

    // O += P(16x64) @ V(64x128), K dim in 2 steps of 32.
#pragma unroll
    for (int ks = 0; ks < 2; ++ks) {
      v16bf a = load_frag_a(&Ps[0][0], 72, qr, ks * 32, lane);
#pragma unroll
      for (int n = 0; n < 8; ++n) {
        v16bf bfrag = load_frag_b(&Vt[0][0], 72, n * 16, ks * 32, lane);
        o[n] = WMMA_BF16(a, bfrag, o[n]);
      }
    }
  }

  // Normalize and scatter to context [B,S,D] (bf16) for the output GEMM.
  const int bq = bh >> 4, hq = bh & 15;
#pragma unroll
  for (int v = 0; v < 8; ++v) {
    const int   sg  = qbase + qr + g * 8 + v;
    const float inv = 1.0f / li[v];
#pragma unroll
    for (int n = 0; n < 8; ++n) {
      const int dg = n * 16 + col;
      Ctx[((size_t)bq * kS + sg) * kD + hq * kHd + dg] = (bf16_t)(o[n][v] * inv);
    }
  }
}

// ---------------------------------------------------------------------------
// Kernel 4: out = Ctx(bf16,[M,D]) @ Wo^T(f32,[D,D]) -> f32 [B,S,D].
// A operand is bf16 in global memory -> async DMA straight into LDS.
// ---------------------------------------------------------------------------
__global__ __launch_bounds__(128)
void gemm_out_kernel(const bf16_t* __restrict__ A, const float* __restrict__ W,
                     float* __restrict__ Y) {
  __shared__ alignas(16) bf16_t As[64][72];
  __shared__ alignas(16) bf16_t Ws[64][72];

  const int tid  = threadIdx.x;
  const int lane = tid & 31;
  const int wave = tid >> 5;
  const int m0 = blockIdx.y * 64;
  const int n0 = blockIdx.x * 64;
  const int wm = (wave >> 1) * 32;
  const int wn = (wave & 1) * 32;

  v8f acc[2][2] = {};
  const int lr = tid >> 1;
  const int lc = (tid & 1) * 32;

  for (int k0 = 0; k0 < kD; k0 += 64) {
    const bf16_t* ap = A + (size_t)(m0 + lr) * kD + k0 + lc;
    const float*  wp = W + (size_t)(n0 + lr) * kD + k0 + lc;
    if (k0 + 64 < kD) __builtin_prefetch(wp + 64, 0, 1);

    // A tile: async global->LDS (already bf16).
    {
      const unsigned long long ag = (unsigned long long)(uintptr_t)ap;
      const unsigned ao = lds_off32(&As[lr][lc]);
#pragma unroll
      for (int c = 0; c < 32; c += 8)
        ASYNC_G2L_B128(ao + (unsigned)(c * 2), ag + (unsigned long long)(c * 2));
    }
    // W tile: f32 -> bf16 conversion through VGPRs.
#pragma unroll
    for (int c = 0; c < 32; c += 8) {
      v8bf wv;
#pragma unroll
      for (int j = 0; j < 8; ++j) wv[j] = (bf16_t)wp[c + j];
      *(v8bf*)&Ws[lr][lc + c] = wv;
    }
    WAIT_ASYNC0();
    __syncthreads();
#pragma unroll
    for (int kk = 0; kk < 64; kk += 32) {
      v16bf a0 = load_frag_a(&As[0][0], 72, wm,      kk, lane);
      v16bf a1 = load_frag_a(&As[0][0], 72, wm + 16, kk, lane);
      v16bf b0 = load_frag_b(&Ws[0][0], 72, wn,      kk, lane);
      v16bf b1 = load_frag_b(&Ws[0][0], 72, wn + 16, kk, lane);
      acc[0][0] = WMMA_BF16(a0, b0, acc[0][0]);
      acc[0][1] = WMMA_BF16(a0, b1, acc[0][1]);
      acc[1][0] = WMMA_BF16(a1, b0, acc[1][0]);
      acc[1][1] = WMMA_BF16(a1, b1, acc[1][1]);
    }
    __syncthreads();
  }

  const int g   = lane >> 4;
  const int col = lane & 15;
#pragma unroll
  for (int i = 0; i < 2; ++i)
#pragma unroll
    for (int j = 0; j < 2; ++j)
#pragma unroll
      for (int v = 0; v < 8; ++v) {
        const int mrow = m0 + wm + i * 16 + g * 8 + v;
        const int ncol = n0 + wn + j * 16 + col;
        Y[(size_t)mrow * kD + ncol] = acc[i][j][v];
      }
}

// ---------------------------------------------------------------------------
// Host launcher.
// ---------------------------------------------------------------------------
extern "C" void kernel_launch(void* const* d_in, const int* in_sizes, int n_in,
                              void* d_out, int out_size, void* d_ws, size_t ws_size,
                              hipStream_t stream) {
  const float* x  = (const float*)d_in[0];
  const float* Wq = (const float*)d_in[1];
  const float* Wk = (const float*)d_in[2];
  const float* Wv = (const float*)d_in[3];
  const float* Wo = (const float*)d_in[4];
  float* out = (float*)d_out;

  const size_t nQ = (size_t)kB * kH * kS * kHd;   // 8,388,608 elems each
  bf16_t* Qb  = (bf16_t*)d_ws;
  bf16_t* Kb  = Qb + nQ;
  bf16_t* Vb  = Kb + nQ;
  bf16_t* Ctx = Vb + nQ;                          // [B,S,D] bf16

  const dim3 gemm_grid(kD / 64, kM / 64);         // (32, 64)
  gemm_proj_kernel<<<gemm_grid, 128, 0, stream>>>(x, Wq, Qb);
  gemm_proj_kernel<<<gemm_grid, 128, 0, stream>>>(x, Wk, Kb);
  gemm_proj_kernel<<<gemm_grid, 128, 0, stream>>>(x, Wv, Vb);

  const size_t rope_threads = (size_t)kB * kH * kS * 64;
  rope_kernel<<<(unsigned)(rope_threads / 256), 256, 0, stream>>>(Qb, Kb);

  attn_kernel<<<dim3(kS / 64, kB * kH), 128, 0, stream>>>(Qb, Kb, Vb, Ctx);

  gemm_out_kernel<<<gemm_grid, 128, 0, stream>>>(Ctx, Wo, out);
}
